// ComplexMoE_39513699123240
// MI455X (gfx1250) — matrix operand order repaired and checked by reading, not compiled, not run on hardware
//
#include <hip/hip_runtime.h>
#include <math.h>

typedef unsigned short u16;
typedef unsigned int   u32;
typedef __attribute__((ext_vector_type(4)))  u32    u32x4;
typedef __attribute__((ext_vector_type(16))) __bf16 v16bf;
typedef __attribute__((ext_vector_type(8)))  float  v8f;

union Frag {
    v16bf v;
    u32x4 q[2];
    u32   w[8];
};

__device__ __forceinline__ u16 f2bf(float f) {
    u32 u = __float_as_uint(f);
    u32 r = u + 0x7fffu + ((u >> 16) & 1u);   // round-to-nearest-even
    return (u16)(r >> 16);
}

__device__ __forceinline__ v8f wmma_bf16(const Frag& a, const Frag& b, v8f c) {
    return __builtin_amdgcn_wmma_f32_16x16x32_bf16(
        /*neg_a=*/false, a.v, /*neg_b=*/false, b.v,
        /*c_mod=*/(short)0, c, /*reuse_a=*/false, /*reuse_b=*/false);
}

// CDNA5 async global->LDS copy (ASYNCcnt tracked). Low 32 bits of a generic
// pointer to __shared__ are the LDS byte offset on AMDGPU.
__device__ __forceinline__ void async_g2l_b128(void* lds_ptr, const void* gptr) {
    asm volatile("global_load_async_to_lds_b128 %0, %1, off"
                 :: "v"((u32)(uintptr_t)lds_ptr), "v"(gptr)
                 : "memory");
}
__device__ __forceinline__ void wait_async0() {
    asm volatile("s_wait_asynccnt 0" ::: "memory");
}

// ---------------------------------------------------------------- utilities
__global__ void cvt_bf16_kernel(const float* __restrict__ src, u16* __restrict__ dst, int n) {
    int i = blockIdx.x * blockDim.x + threadIdx.x;
    if (i < n) dst[i] = f2bf(src[i]);
}

__global__ void zero_kernel(float* __restrict__ dst, int n) {
    int i = blockIdx.x * blockDim.x + threadIdx.x;
    if (i < n) dst[i] = 0.0f;
}

// ---------------------------------------------------------------- router
// one wave (32 lanes) per token; E=4, top-2 softmax
__global__ void router_kernel(const float* __restrict__ xr, const float* __restrict__ xi,
                              const float* __restrict__ rw, const float* __restrict__ rb,
                              float* __restrict__ wfull) {
    int lane = threadIdx.x & 31;
    int wv   = threadIdx.x >> 5;
    int t    = blockIdx.x * 8 + wv;
    const float* pxr = xr + (size_t)t * 512;
    const float* pxi = xi + (size_t)t * 512;
    float a0 = 0.f, a1 = 0.f, a2 = 0.f, a3 = 0.f;
    for (int k = lane; k < 512; k += 32) {
        float vr = pxr[k], vi = pxi[k];
        a0 += vr * rw[0 * 1024 + k] + vi * rw[0 * 1024 + 512 + k];
        a1 += vr * rw[1 * 1024 + k] + vi * rw[1 * 1024 + 512 + k];
        a2 += vr * rw[2 * 1024 + k] + vi * rw[2 * 1024 + 512 + k];
        a3 += vr * rw[3 * 1024 + k] + vi * rw[3 * 1024 + 512 + k];
    }
    for (int off = 16; off > 0; off >>= 1) {
        a0 += __shfl_xor(a0, off, 32);
        a1 += __shfl_xor(a1, off, 32);
        a2 += __shfl_xor(a2, off, 32);
        a3 += __shfl_xor(a3, off, 32);
    }
    if (lane == 0) {
        float lg[4] = {a0 + rb[0], a1 + rb[1], a2 + rb[2], a3 + rb[3]};
        int i1 = 0; float m1 = lg[0];
        for (int q = 1; q < 4; ++q) if (lg[q] > m1) { m1 = lg[q]; i1 = q; }
        int i2 = -1; float m2 = -3.4e38f;
        for (int q = 0; q < 4; ++q) if (q != i1 && lg[q] > m2) { m2 = lg[q]; i2 = q; }
        float e2 = __expf(m2 - m1);
        float inv = 1.0f / (1.0f + e2);
        float w[4] = {0.f, 0.f, 0.f, 0.f};
        w[i1] = inv;
        w[i2] = e2 * inv;
        wfull[t * 4 + 0] = w[0];
        wfull[t * 4 + 1] = w[1];
        wfull[t * 4 + 2] = w[2];
        wfull[t * 4 + 3] = w[3];
    }
}

// ---------------------------------------------------------------- up projection (per expert)
// grid: (DH/32, N/64); block: 256 = 8 waves (4 M-subtiles x 2 N-subtiles)
// K staged in chunks of 64 via async global->LDS; 16 WMMAs per stage.
__global__ __launch_bounds__(256, 1) void up_kernel(
    const u16* __restrict__ xr16, const u16* __restrict__ xi16,
    const u16* __restrict__ wugr, const u16* __restrict__ wugi,
    const u16* __restrict__ wuvr, const u16* __restrict__ wuvi,
    const float* __restrict__ ug_br, const float* __restrict__ ug_bi,
    const float* __restrict__ uv_br, const float* __restrict__ uv_bi,
    u16* __restrict__ hr16, u16* __restrict__ hi16) {
    const int tid  = threadIdx.x;
    const int lane = tid & 31;
    const int wv   = tid >> 5;
    const int wm   = wv & 3;
    const int wn   = wv >> 2;
    const int l16  = lane & 15;
    const int hl   = lane >> 4;
    const int m0   = blockIdx.y * 64;
    const int n0   = blockIdx.x * 32;

    __shared__ __align__(16) u16 Ar[64][72];
    __shared__ __align__(16) u16 Ai[64][72];
    __shared__ __align__(16) u16 Bs[4][32][72];

    v8f agr = {}, agi = {}, avr = {}, avi = {};

    // staging assignments
    const int row_a = tid >> 2;          // 0..63
    const int col_a = (tid & 3) << 4;    // 0,16,32,48
    const int row_b = tid >> 3;          // 0..31
    const int col_b = (tid & 7) << 3;    // 0..56

    const u16* gxr = xr16 + (size_t)(m0 + row_a) * 512 + col_a;
    const u16* gxi = xi16 + (size_t)(m0 + row_a) * 512 + col_a;
    const u16* gb0 = wugr + (size_t)(n0 + row_b) * 512 + col_b;
    const u16* gb1 = wugi + (size_t)(n0 + row_b) * 512 + col_b;
    const u16* gb2 = wuvr + (size_t)(n0 + row_b) * 512 + col_b;
    const u16* gb3 = wuvi + (size_t)(n0 + row_b) * 512 + col_b;

    for (int k0 = 0; k0 < 512; k0 += 64) {
        // stage A (xr, xi): 64x64 each; 32B per thread per matrix (2 x b128)
        async_g2l_b128(&Ar[row_a][col_a],     gxr + k0);
        async_g2l_b128(&Ar[row_a][col_a + 8], gxr + k0 + 8);
        async_g2l_b128(&Ai[row_a][col_a],     gxi + k0);
        async_g2l_b128(&Ai[row_a][col_a + 8], gxi + k0 + 8);
        // stage B (ug_r, ug_i, uv_r, uv_i): 32x64 each; 16B per thread per matrix
        async_g2l_b128(&Bs[0][row_b][col_b], gb0 + k0);
        async_g2l_b128(&Bs[1][row_b][col_b], gb1 + k0);
        async_g2l_b128(&Bs[2][row_b][col_b], gb2 + k0);
        async_g2l_b128(&Bs[3][row_b][col_b], gb3 + k0);
        wait_async0();
        __syncthreads();

        const int am = wm * 16 + l16;
        const int bo = wn * 16 + l16;
#pragma unroll
        for (int ks = 0; ks < 64; ks += 32) {
            Frag far, fai, fnai, bgr, bgi, bvr, bvi;
            far.q[0] = *(const u32x4*)&Ar[am][ks + hl * 16];
            far.q[1] = *(const u32x4*)&Ar[am][ks + hl * 16 + 8];
            fai.q[0] = *(const u32x4*)&Ai[am][ks + hl * 16];
            fai.q[1] = *(const u32x4*)&Ai[am][ks + hl * 16 + 8];
#pragma unroll
            for (int j = 0; j < 8; ++j) fnai.w[j] = fai.w[j] ^ 0x80008000u;
            bgr.q[0] = *(const u32x4*)&Bs[0][bo][ks + hl * 16];
            bgr.q[1] = *(const u32x4*)&Bs[0][bo][ks + hl * 16 + 8];
            bgi.q[0] = *(const u32x4*)&Bs[1][bo][ks + hl * 16];
            bgi.q[1] = *(const u32x4*)&Bs[1][bo][ks + hl * 16 + 8];
            bvr.q[0] = *(const u32x4*)&Bs[2][bo][ks + hl * 16];
            bvr.q[1] = *(const u32x4*)&Bs[2][bo][ks + hl * 16 + 8];
            bvi.q[0] = *(const u32x4*)&Bs[3][bo][ks + hl * 16];
            bvi.q[1] = *(const u32x4*)&Bs[3][bo][ks + hl * 16 + 8];

            agr = wmma_bf16(far,  bgr, agr);  agr = wmma_bf16(fnai, bgi, agr);
            agi = wmma_bf16(fai,  bgr, agi);  agi = wmma_bf16(far,  bgi, agi);
            avr = wmma_bf16(far,  bvr, avr);  avr = wmma_bf16(fnai, bvi, avr);
            avi = wmma_bf16(fai,  bvr, avi);  avi = wmma_bf16(far,  bvi, avi);
        }
        __syncthreads();
    }

    const int dh = n0 + wn * 16 + l16;
    const float bgr_s = ug_br[dh], bgi_s = ug_bi[dh];
    const float bvr_s = uv_br[dh], bvi_s = uv_bi[dh];
#pragma unroll
    for (int v = 0; v < 8; ++v) {
        int token = m0 + wm * 16 + v + hl * 8;
        float gr = agr[v] + bgr_s;
        float gi = agi[v] + bgi_s;
        float mag = sqrtf(gr * gr + gi * gi + 1e-8f);
        float gate = mag / (1.0f + __expf(-mag));   // silu
        float hr = gate * (avr[v] + bvr_s);
        float hi = gate * (avi[v] + bvi_s);
        hr16[(size_t)token * 1024 + dh] = f2bf(hr);
        hi16[(size_t)token * 1024 + dh] = f2bf(hi);
    }
}

// ---------------------------------------------------------------- down projection (per expert)
// grid: (D/64, N/64); block: 256 = 8 waves (4 M x 2 N), each wave computes 16x32.
// K staged in chunks of 64; 16 WMMAs per stage; epilogue scales by router weight.
__global__ __launch_bounds__(256, 1) void down_kernel(
    const u16* __restrict__ hr16, const u16* __restrict__ hi16,
    const u16* __restrict__ wdr, const u16* __restrict__ wdi,
    const float* __restrict__ dn_br, const float* __restrict__ dn_bi,
    const float* __restrict__ wfull, int e,
    float* __restrict__ out_r, float* __restrict__ out_i) {
    const int tid  = threadIdx.x;
    const int lane = tid & 31;
    const int wv   = tid >> 5;
    const int wm   = wv & 3;
    const int wn   = wv >> 2;
    const int l16  = lane & 15;
    const int hl   = lane >> 4;
    const int m0   = blockIdx.y * 64;
    const int n0   = blockIdx.x * 64;

    __shared__ __align__(16) u16 Ahr[64][72];
    __shared__ __align__(16) u16 Ahi[64][72];
    __shared__ __align__(16) u16 Bs[2][64][72];

    v8f yr0 = {}, yi0 = {}, yr1 = {}, yi1 = {};

    const int row_a = tid >> 2;          // 0..63
    const int col_a = (tid & 3) << 4;    // 0,16,32,48

    const u16* ghr = hr16 + (size_t)(m0 + row_a) * 1024 + col_a;
    const u16* ghi = hi16 + (size_t)(m0 + row_a) * 1024 + col_a;
    const u16* gw0 = wdr + (size_t)(n0 + row_a) * 1024 + col_a;
    const u16* gw1 = wdi + (size_t)(n0 + row_a) * 1024 + col_a;

    for (int k0 = 0; k0 < 1024; k0 += 64) {
        async_g2l_b128(&Ahr[row_a][col_a],     ghr + k0);
        async_g2l_b128(&Ahr[row_a][col_a + 8], ghr + k0 + 8);
        async_g2l_b128(&Ahi[row_a][col_a],     ghi + k0);
        async_g2l_b128(&Ahi[row_a][col_a + 8], ghi + k0 + 8);
        async_g2l_b128(&Bs[0][row_a][col_a],     gw0 + k0);
        async_g2l_b128(&Bs[0][row_a][col_a + 8], gw0 + k0 + 8);
        async_g2l_b128(&Bs[1][row_a][col_a],     gw1 + k0);
        async_g2l_b128(&Bs[1][row_a][col_a + 8], gw1 + k0 + 8);
        wait_async0();
        __syncthreads();

        const int am  = wm * 16 + l16;
        const int bo0 = wn * 32 + l16;
        const int bo1 = bo0 + 16;
#pragma unroll
        for (int ks = 0; ks < 64; ks += 32) {
            Frag fhr, fhi, fnhi, bwr0, bwi0, bwr1, bwi1;
            fhr.q[0] = *(const u32x4*)&Ahr[am][ks + hl * 16];
            fhr.q[1] = *(const u32x4*)&Ahr[am][ks + hl * 16 + 8];
            fhi.q[0] = *(const u32x4*)&Ahi[am][ks + hl * 16];
            fhi.q[1] = *(const u32x4*)&Ahi[am][ks + hl * 16 + 8];
#pragma unroll
            for (int j = 0; j < 8; ++j) fnhi.w[j] = fhi.w[j] ^ 0x80008000u;
            bwr0.q[0] = *(const u32x4*)&Bs[0][bo0][ks + hl * 16];
            bwr0.q[1] = *(const u32x4*)&Bs[0][bo0][ks + hl * 16 + 8];
            bwi0.q[0] = *(const u32x4*)&Bs[1][bo0][ks + hl * 16];
            bwi0.q[1] = *(const u32x4*)&Bs[1][bo0][ks + hl * 16 + 8];
            bwr1.q[0] = *(const u32x4*)&Bs[0][bo1][ks + hl * 16];
            bwr1.q[1] = *(const u32x4*)&Bs[0][bo1][ks + hl * 16 + 8];
            bwi1.q[0] = *(const u32x4*)&Bs[1][bo1][ks + hl * 16];
            bwi1.q[1] = *(const u32x4*)&Bs[1][bo1][ks + hl * 16 + 8];

            yr0 = wmma_bf16(fhr,  bwr0, yr0);  yr0 = wmma_bf16(fnhi, bwi0, yr0);
            yi0 = wmma_bf16(fhi,  bwr0, yi0);  yi0 = wmma_bf16(fhr,  bwi0, yi0);
            yr1 = wmma_bf16(fhr,  bwr1, yr1);  yr1 = wmma_bf16(fnhi, bwi1, yr1);
            yi1 = wmma_bf16(fhi,  bwr1, yi1);  yi1 = wmma_bf16(fhr,  bwi1, yi1);
        }
        __syncthreads();
    }

    const int col0 = n0 + wn * 32 + l16;
    const int col1 = col0 + 16;
    const float br0 = dn_br[col0], bi0 = dn_bi[col0];
    const float br1 = dn_br[col1], bi1 = dn_bi[col1];
#pragma unroll
    for (int v = 0; v < 8; ++v) {
        int token = m0 + wm * 16 + v + hl * 8;
        float we = wfull[token * 4 + e];
        size_t idx = (size_t)token * 512;
        out_r[idx + col0] += (yr0[v] + br0) * we;
        out_i[idx + col0] += (yi0[v] + bi0) * we;
        out_r[idx + col1] += (yr1[v] + br1) * we;
        out_i[idx + col1] += (yi1[v] + bi1) * we;
    }
}

// ---------------------------------------------------------------- launch
extern "C" void kernel_launch(void* const* d_in, const int* in_sizes, int n_in,
                              void* d_out, int out_size, void* d_ws, size_t ws_size,
                              hipStream_t stream) {
    (void)in_sizes; (void)n_in; (void)out_size; (void)ws_size;
    const int NTOK = 8192, D = 512, DH = 1024, E = 4;

    const float* x_r      = (const float*)d_in[0];
    const float* x_i      = (const float*)d_in[1];
    const float* router_w = (const float*)d_in[2];
    const float* router_b = (const float*)d_in[3];
    const float* ug_wr    = (const float*)d_in[4];
    const float* ug_wi    = (const float*)d_in[5];
    const float* ug_br    = (const float*)d_in[6];
    const float* ug_bi    = (const float*)d_in[7];
    const float* uv_wr    = (const float*)d_in[8];
    const float* uv_wi    = (const float*)d_in[9];
    const float* uv_br    = (const float*)d_in[10];
    const float* uv_bi    = (const float*)d_in[11];
    const float* dn_wr    = (const float*)d_in[12];
    const float* dn_wi    = (const float*)d_in[13];
    const float* dn_br    = (const float*)d_in[14];
    const float* dn_bi    = (const float*)d_in[15];

    float* out_r = (float*)d_out;
    float* out_i = out_r + (size_t)NTOK * D;

    size_t off = 0;
    auto walloc = [&](size_t bytes) -> void* {
        void* p = (char*)d_ws + off;
        off += (bytes + 255) & ~(size_t)255;
        return p;
    };
    u16* xr16   = (u16*)walloc((size_t)NTOK * D * 2);
    u16* xi16   = (u16*)walloc((size_t)NTOK * D * 2);
    u16* wugr16 = (u16*)walloc((size_t)E * DH * D * 2);
    u16* wugi16 = (u16*)walloc((size_t)E * DH * D * 2);
    u16* wuvr16 = (u16*)walloc((size_t)E * DH * D * 2);
    u16* wuvi16 = (u16*)walloc((size_t)E * DH * D * 2);
    u16* wdnr16 = (u16*)walloc((size_t)E * D * DH * 2);
    u16* wdni16 = (u16*)walloc((size_t)E * D * DH * 2);
    float* wfull = (float*)walloc((size_t)NTOK * E * 4);
    u16* hr16   = (u16*)walloc((size_t)NTOK * DH * 2);
    u16* hi16   = (u16*)walloc((size_t)NTOK * DH * 2);

    // f32 -> bf16 conversions
    const int nx = NTOK * D;
    const int nw = E * DH * D;
    cvt_bf16_kernel<<<(nx + 255) / 256, 256, 0, stream>>>(x_r,   xr16,   nx);
    cvt_bf16_kernel<<<(nx + 255) / 256, 256, 0, stream>>>(x_i,   xi16,   nx);
    cvt_bf16_kernel<<<(nw + 255) / 256, 256, 0, stream>>>(ug_wr, wugr16, nw);
    cvt_bf16_kernel<<<(nw + 255) / 256, 256, 0, stream>>>(ug_wi, wugi16, nw);
    cvt_bf16_kernel<<<(nw + 255) / 256, 256, 0, stream>>>(uv_wr, wuvr16, nw);
    cvt_bf16_kernel<<<(nw + 255) / 256, 256, 0, stream>>>(uv_wi, wuvi16, nw);
    cvt_bf16_kernel<<<(nw + 255) / 256, 256, 0, stream>>>(dn_wr, wdnr16, nw);
    cvt_bf16_kernel<<<(nw + 255) / 256, 256, 0, stream>>>(dn_wi, wdni16, nw);

    // zero output accumulators (out_r and out_i are contiguous)
    const int nout = 2 * NTOK * D;
    zero_kernel<<<(nout + 255) / 256, 256, 0, stream>>>(out_r, nout);

    // router weights
    router_kernel<<<NTOK / 8, 256, 0, stream>>>(x_r, x_i, router_w, router_b, wfull);

    // experts
    for (int e = 0; e < E; ++e) {
        const size_t wo = (size_t)e * DH * D;
        up_kernel<<<dim3(DH / 32, NTOK / 64), 256, 0, stream>>>(
            xr16, xi16,
            wugr16 + wo, wugi16 + wo, wuvr16 + wo, wuvi16 + wo,
            ug_br + (size_t)e * DH, ug_bi + (size_t)e * DH,
            uv_br + (size_t)e * DH, uv_bi + (size_t)e * DH,
            hr16, hi16);
        down_kernel<<<dim3(D / 64, NTOK / 64), 256, 0, stream>>>(
            hr16, hi16,
            wdnr16 + wo, wdni16 + wo,
            dn_br + (size_t)e * D, dn_bi + (size_t)e * D,
            wfull, e, out_r, out_i);
    }
}
